// MHSA_3908420240056
// MI455X (gfx1250) — compile-verified, matrix operand order checked by reading, not run on hardware
//
#include <hip/hip_runtime.h>
#include <hip/hip_bf16.h>

// ---------------------------------------------------------------------------
// MHSA (B=2, S=2048, H=8, head_dim=512) fused pipeline for gfx1250 (MI455X).
//  * f32 -> f16 casts
//  * WMMA (v_wmma_f32_16x16x32_f16) for all GEMMs
//  * flash attention: K / V^T tiles staged into LDS by the Tensor Data Mover
//    (tensor_load_to_lds, TENSORcnt double-buffered), score partials reduced
//    via LDS (no [S,S] materialization)
//  * out-proj + residual + LayerNorm epilogue
// ---------------------------------------------------------------------------

typedef __attribute__((ext_vector_type(16))) _Float16 v16h;
typedef __attribute__((ext_vector_type(8)))  float    v8f;
typedef __attribute__((ext_vector_type(4)))  unsigned int v4u;
typedef __attribute__((ext_vector_type(4)))  int      v4i_t;
typedef __attribute__((ext_vector_type(8)))  int      v8i_t;

#define DEV __device__ __forceinline__

constexpr int   HH   = 8;
constexpr int   DIN  = 512;
constexpr int   DOUT = 512;
constexpr int   BB   = 2;
constexpr int   SS   = 2048;
constexpr int   HD   = HH * DOUT;             // 4096
constexpr float SCALE = 0.04419417382415922f; // 1/sqrt(512)
constexpr float LNEPS = 1e-5f;

#if __has_builtin(__builtin_amdgcn_tensor_load_to_lds) && \
    __has_builtin(__builtin_amdgcn_s_wait_tensorcnt)
#define USE_TDM 1
#else
#define USE_TDM 0
#endif

DEV v8f wmma16(v16h a, v16h b, v8f c) {
  // D = A(16x32,f16) * B(32x16,f16) + C(16x16,f32)
  return __builtin_amdgcn_wmma_f32_16x16x32_f16(
      /*neg_a=*/false, a, /*neg_b=*/false, b,
      /*c_mod=*/(short)0, c, /*reuse_a=*/false, /*reuse_b=*/false);
}

DEV v16h ldg16(const _Float16* p) { return *(const v16h*)p; }
DEV v16h lds16(const _Float16* p) { return *(const v16h*)p; }

#if USE_TDM
// ---------------------------------------------------------------------------
// Tensor Data Mover: 2D tile (16-bit elements) global -> LDS.
// D# layout per CDNA5 ISA (groups 0/1; groups 2/3 zero,
// tile_dim2/3/4 = 0 => unused).
// ---------------------------------------------------------------------------
DEV void tdm_load_2d_f16(unsigned int lds_addr, const void* gptr,
                         unsigned int tensor_d0, unsigned int tensor_d1,
                         unsigned int tile_d0,   unsigned int tile_d1,
                         unsigned int stride0) {
  const unsigned long long ga = (unsigned long long)(uintptr_t)gptr;
  v4u g0 = { 1u,                                   // count=1, user mode
             lds_addr,                             // lds_addr [63:32]
             (unsigned int)ga,                     // global_addr [95:64]
             ((unsigned int)(ga >> 32) & 0x01ffffffu) | (2u << 30) }; // [120:96], type=2
  v8i_t g1 = { (int)(1u << 16),                                  // data_size=1 (2 bytes)
               (int)((tensor_d0 & 0xffffu) << 16),               // tensor_dim0 lo
               (int)((tensor_d0 >> 16) | ((tensor_d1 & 0xffffu) << 16)),
               (int)((tensor_d1 >> 16) | ((tile_d0 & 0xffffu) << 16)),
               (int)(tile_d1 & 0xffffu),                         // tile_dim1, tile_dim2=0
               (int)stride0,                                     // tensor_dim0_stride lo
               0, 0 };
  v4i_t gz = { 0, 0, 0, 0 };
#if defined(__clang_major__) && __clang_major__ >= 23
  v8i_t gz8 = { 0, 0, 0, 0, 0, 0, 0, 0 };
  __builtin_amdgcn_tensor_load_to_lds(g0, g1, gz, gz, gz8, 0);
#else
  __builtin_amdgcn_tensor_load_to_lds(g0, g1, gz, gz, 0);
#endif
}
DEV unsigned int lds_off(const void* p) { return (unsigned int)(uintptr_t)p; }
#endif

// ---------------------------------------------------------------------------
// Kernel 1: f32 -> f16 cast (grid-stride)
// ---------------------------------------------------------------------------
__global__ void cast_f32_f16(const float* __restrict__ src,
                             _Float16* __restrict__ dst, int n) {
  int i = blockIdx.x * blockDim.x + threadIdx.x;
  int stride = gridDim.x * blockDim.x;
  for (; i < n; i += stride) dst[i] = (_Float16)src[i];
}

// ---------------------------------------------------------------------------
// Kernel 2: per-head Q,K,V projections.
//   Q[bh,s,o] = (D[b] @ Wq[h]^T) * SCALE, K likewise, Vt[bh,o,s] transposed.
// grid = (S/128, B*H), block = 256 (8 waves); wave owns a 16-row strip and
// caches all 16 A-fragments (16x512 of D) in VGPRs, streams weight B-frags.
// ---------------------------------------------------------------------------
__global__ __launch_bounds__(256, 1)
void qkv_proj(const _Float16* __restrict__ Dh,
              const _Float16* __restrict__ Wqh,
              const _Float16* __restrict__ Wkh,
              const _Float16* __restrict__ Wvh,
              _Float16* __restrict__ Qo,
              _Float16* __restrict__ Ko,
              _Float16* __restrict__ Vt) {
  const int lane  = threadIdx.x & 31;
  const int wave  = threadIdx.x >> 5;
  const int bh    = blockIdx.y;          // 0..15
  const int b     = bh >> 3;
  const int h     = bh & 7;
  const int strip = blockIdx.x * 8 + wave;   // 0..127
  const int row0  = strip * 16;
  const int mrow  = lane & 15;
  const int khalf = (lane >> 4) * 16;    // lanes 16..31 hold K+16..K+31
  const int rtop  = (lane >> 4) * 8;     // C-tile rows: v + rtop

  // Cache A fragments: full 16x512 strip of D (128 VGPRs).
  const _Float16* db = Dh + ((size_t)(b * SS + row0 + mrow)) * DIN + khalf;
  v16h a[16];
#pragma unroll
  for (int k = 0; k < 16; ++k) a[k] = ldg16(db + k * 32);

  const _Float16* wq = Wqh + (size_t)h * DOUT * DIN;
  const _Float16* wk = Wkh + (size_t)h * DOUT * DIN;
  const _Float16* wv = Wvh + (size_t)h * DOUT * DIN;
  const size_t qkbase = (size_t)bh * SS * DOUT;
  const size_t vtbase = (size_t)bh * DOUT * SS;

  for (int nt = 0; nt < 32; ++nt) {
    const int col = nt * 16 + mrow;
    const size_t wro = (size_t)col * DIN + khalf;
    if (nt + 1 < 32)   // warm next weight tile -> global_prefetch_b8
      __builtin_prefetch(wq + wro + DIN * 16, 0, 2);

    v8f acc = {};
#pragma unroll
    for (int k = 0; k < 16; ++k) acc = wmma16(a[k], ldg16(wq + wro + k * 32), acc);
#pragma unroll
    for (int v = 0; v < 8; ++v) {
      int grow = row0 + v + rtop;
      Qo[qkbase + (size_t)grow * DOUT + col] = (_Float16)(acc[v] * SCALE);
    }

    acc = (v8f){};
#pragma unroll
    for (int k = 0; k < 16; ++k) acc = wmma16(a[k], ldg16(wk + wro + k * 32), acc);
#pragma unroll
    for (int v = 0; v < 8; ++v) {
      int grow = row0 + v + rtop;
      Ko[qkbase + (size_t)grow * DOUT + col] = (_Float16)acc[v];
    }

    acc = (v8f){};
#pragma unroll
    for (int k = 0; k < 16; ++k) acc = wmma16(a[k], ldg16(wv + wro + k * 32), acc);
#pragma unroll
    for (int v = 0; v < 8; ++v) {
      int grow = row0 + v + rtop;
      Vt[vtbase + (size_t)col * SS + grow] = (_Float16)acc[v];  // transposed
    }
  }
}

// ---------------------------------------------------------------------------
// Kernel 3: flash attention per (b,h).
// grid = (S/16, B*H), block = 256 (8 waves). Br=16 queries, Bc=32 keys/step.
// K tile (32x512) and V^T tile (512x32, strided) are DMA'd into LDS by the
// TDM (double-buffered, TENSORcnt), fragments come from LDS. The d=512 score
// reduction is split across 8 waves; partials land in per-wave LDS slices and
// are summed by all threads. Wave 0 runs the online softmax.
// Each wave owns a 16x64 slice of O (4 f32 accumulator tiles).
// LDS: 64K (K) + 64K (V) + 16K + 2K + 1K + 128B ~= 147 KB (<320 KB WGP).
// ---------------------------------------------------------------------------
constexpr int TK = 32 * 512;   // K tile elements
constexpr int TV = 512 * 32;   // V tile elements

__global__ __launch_bounds__(256, 1)
void attn_flash(const _Float16* __restrict__ Q,
                const _Float16* __restrict__ K,
                const _Float16* __restrict__ Vt,
                _Float16* __restrict__ Xc) {
  __shared__ __align__(128) _Float16 sK[2 * TK];     // 64 KB, double-buffered
  __shared__ __align__(128) _Float16 sV[2 * TV];     // 64 KB, double-buffered
  __shared__ __align__(128) float    sS[8 * 512];    // 16 KB per-wave partials
  __shared__ __align__(128) float    sR[512];        // reduced 16x32 scores
  __shared__ __align__(128) _Float16 sP[512];        // probabilities (f16)
  __shared__ __align__(64)  float    sAlpha[16];
  __shared__ __align__(64)  float    sL[16];

  const int lane  = threadIdx.x & 31;
  const int wave  = threadIdx.x >> 5;
  const int bh    = blockIdx.y;
  const int b     = bh >> 3;
  const int h     = bh & 7;
  const int i0    = blockIdx.x * 16;     // query tile start
  const int mrow  = lane & 15;
  const int khalf = (lane >> 4) * 16;
  const int rtop  = (lane >> 4) * 8;

  const _Float16* Qb = Q  + (size_t)bh * SS * DOUT;
  const _Float16* Kb = K  + (size_t)bh * SS * DOUT;
  const _Float16* Vb = Vt + (size_t)bh * DOUT * SS;

  // This wave's Q A-fragments for k-slice [wave*64, wave*64+64): 2 frags.
  v16h qa[2];
#pragma unroll
  for (int s = 0; s < 2; ++s)
    qa[s] = ldg16(Qb + (size_t)(i0 + mrow) * DOUT + wave * 64 + s * 32 + khalf);

  v8f o[4] = {};
  float m_r = -3.0e38f;   // running max (wave 0, row = lane&15)
  float l_r = 0.0f;       // running sum

  const int nJ = SS / 32;  // 64 key tiles

#if USE_TDM
  if (wave == 0) {   // prologue: DMA tile 0 into buffer 0
    tdm_load_2d_f16(lds_off(&sK[0]), Kb, /*t0=*/DOUT, /*t1=*/SS,
                    /*tile0=*/DOUT, /*tile1=*/32, /*stride0=*/DOUT);
    tdm_load_2d_f16(lds_off(&sV[0]), Vb, /*t0=*/SS, /*t1=*/DOUT,
                    /*tile0=*/32, /*tile1=*/DOUT, /*stride0=*/SS);
  }
#endif

  for (int j = 0; j < nJ; ++j) {
    const int key0 = j * 32;
    const int cur  = j & 1;

#if USE_TDM
    if (wave == 0) __builtin_amdgcn_s_wait_tensorcnt(0);  // tile j resident
    __syncthreads();                                      // publish to all waves
    if (wave == 0 && j + 1 < nJ) {                        // DMA tile j+1, overlap
      const int k1 = key0 + 32;
      tdm_load_2d_f16(lds_off(&sK[(cur ^ 1) * TK]), Kb + (size_t)k1 * DOUT,
                      DOUT, SS, DOUT, 32, DOUT);
      tdm_load_2d_f16(lds_off(&sV[(cur ^ 1) * TV]), Vb + k1,
                      SS, DOUT, 32, DOUT, SS);
    }
#else
    {   // synchronous staging fallback (vectorized, double-buffered)
      const _Float16* kg = Kb + (size_t)key0 * DOUT;
      _Float16* kd = sK + cur * TK;
      for (int i = threadIdx.x * 8; i < TK; i += 256 * 8)
        *(uint4*)(kd + i) = *(const uint4*)(kg + i);
      _Float16* vd = sV + cur * TV;
      for (int r = threadIdx.x; r < DOUT; r += 256) {
        const _Float16* vg = Vb + (size_t)r * SS + key0;
        *(uint4*)(vd + r * 32)      = *(const uint4*)(vg);
        *(uint4*)(vd + r * 32 + 8)  = *(const uint4*)(vg + 8);
        *(uint4*)(vd + r * 32 + 16) = *(const uint4*)(vg + 16);
        *(uint4*)(vd + r * 32 + 24) = *(const uint4*)(vg + 24);
      }
      __syncthreads();
    }
#endif

    // --- partial scores for this wave's k-slice: two 16x16 tiles ---
    const _Float16* kt = sK + cur * TK;
#pragma unroll
    for (int nsub = 0; nsub < 2; ++nsub) {
      v8f sc = {};
#pragma unroll
      for (int s = 0; s < 2; ++s) {
        v16h kb = lds16(kt + (nsub * 16 + mrow) * DOUT + wave * 64 + s * 32 + khalf);
        sc = wmma16(qa[s], kb, sc);
      }
      const int col = nsub * 16 + mrow;
#pragma unroll
      for (int v = 0; v < 8; ++v)
        sS[wave * 512 + (v + rtop) * 32 + col] = sc[v];   // private slice, no atomics
    }
    __syncthreads();

    // --- distributed 8-way reduction of per-wave partials ---
    for (int e = threadIdx.x; e < 512; e += 256) {
      float acc = 0.0f;
#pragma unroll
      for (int w = 0; w < 8; ++w) acc += sS[w * 512 + e];
      sR[e] = acc;
    }
    __syncthreads();

    // --- wave 0: online softmax over the reduced 16x32 score tile ---
    if (wave == 0) {
      float mx = m_r;
#pragma unroll
      for (int kk = 0; kk < 32; ++kk) mx = fmaxf(mx, sR[mrow * 32 + kk]);
      float alpha = __expf(m_r - mx);
      float sum = 0.0f;
#pragma unroll
      for (int kk = 0; kk < 32; ++kk) {
        float p = __expf(sR[mrow * 32 + kk] - mx);
        sum += p;
        if (lane < 16) sP[mrow * 32 + kk] = (_Float16)p;
      }
      l_r = l_r * alpha + sum;
      m_r = mx;
      if (lane < 16) sAlpha[mrow] = alpha;
    }
    __syncthreads();

    // --- rescale O accumulators ---
#pragma unroll
    for (int v = 0; v < 8; ++v) {
      float al = sAlpha[v + rtop];
      o[0][v] *= al; o[1][v] *= al; o[2][v] *= al; o[3][v] *= al;
    }

    // --- O += P(16x32) x V(32x16) per column tile (V from staged LDS) ---
    v16h pa = lds16(sP + mrow * 32 + khalf);
    const _Float16* vt = sV + cur * TV;
#pragma unroll
    for (int t = 0; t < 4; ++t) {
      v16h vb = lds16(vt + (wave * 64 + t * 16 + mrow) * 32 + khalf);
      o[t] = wmma16(pa, vb, o[t]);
    }
    // next iteration's first barrier orders buffer reuse
  }

  if (wave == 0 && lane < 16) sL[mrow] = l_r;
  __syncthreads();

  // normalize and write to Xc[b, s, h*512 + col]
#pragma unroll
  for (int t = 0; t < 4; ++t) {
    const int col = h * DOUT + wave * 64 + t * 16 + mrow;
#pragma unroll
    for (int v = 0; v < 8; ++v) {
      const int row = i0 + v + rtop;
      const float inv = 1.0f / sL[v + rtop];
      Xc[((size_t)(b * SS + row)) * HD + col] = (_Float16)(o[t][v] * inv);
    }
  }
}

// ---------------------------------------------------------------------------
// Kernel 4: res = Xc @ Wo^T + bo; y = D + res; out = LayerNorm(y)*gamma+beta
// grid = B*S/16, block = 256 (8 waves); wave owns a 16x64 slice (K=4096).
// ---------------------------------------------------------------------------
__global__ __launch_bounds__(256, 1)
void oproj_ln(const _Float16* __restrict__ Xc,
              const _Float16* __restrict__ Woh,
              const float* __restrict__ Din,
              const float* __restrict__ bo,
              const float* __restrict__ gamma,
              const float* __restrict__ beta,
              float* __restrict__ out) {
  __shared__ __align__(128) float sY[16 * 512];  // 32 KB

  const int lane  = threadIdx.x & 31;
  const int wave  = threadIdx.x >> 5;
  const int row0  = blockIdx.x * 16;     // rows of flattened [B*S]
  const int mrow  = lane & 15;
  const int khalf = (lane >> 4) * 16;
  const int rtop  = (lane >> 4) * 8;

  v8f acc[4] = {};
  const _Float16* xrow = Xc + (size_t)(row0 + mrow) * HD + khalf;
  for (int k = 0; k < 128; ++k) {        // K = 4096 = 128 * 32
    v16h a = ldg16(xrow + k * 32);
#pragma unroll
    for (int t = 0; t < 4; ++t) {
      v16h bf = ldg16(Woh + (size_t)(wave * 64 + t * 16 + mrow) * HD + k * 32 + khalf);
      acc[t] = wmma16(a, bf, acc[t]);
    }
  }

  // bias + residual into LDS (f32)
#pragma unroll
  for (int t = 0; t < 4; ++t) {
    const int col = wave * 64 + t * 16 + mrow;
    const float bias = bo[col];
#pragma unroll
    for (int v = 0; v < 8; ++v) {
      const int lrow = v + rtop;
      float y = acc[t][v] + bias + Din[(size_t)(row0 + lrow) * DOUT + col];
      sY[lrow * 512 + col] = y;
    }
  }
  __syncthreads();

  // LayerNorm: wave w handles rows {w, w+8}; wave32 butterfly reduction.
  for (int rr = wave; rr < 16; rr += 8) {
    float sum = 0.0f, sq = 0.0f;
#pragma unroll
    for (int t = 0; t < 16; ++t) {
      float y = sY[rr * 512 + lane + t * 32];
      sum += y; sq += y * y;
    }
#pragma unroll
    for (int off = 16; off >= 1; off >>= 1) {
      sum += __shfl_xor(sum, off);
      sq  += __shfl_xor(sq, off);
    }
    const float mean = sum * (1.0f / 512.0f);
    const float var  = sq * (1.0f / 512.0f) - mean * mean;
    const float rstd = rsqrtf(var + LNEPS);
    const size_t obase = (size_t)(row0 + rr) * DOUT;
#pragma unroll
    for (int t = 0; t < 16; ++t) {
      const int c = lane + t * 32;
      const float y = sY[rr * 512 + c];
      out[obase + c] = (y - mean) * rstd * gamma[c] + beta[c];
    }
  }
}

// ---------------------------------------------------------------------------
// Launch: casts -> qkv_proj -> attn_flash -> oproj_ln (all on `stream`).
// Workspace (f16): Dh | Wqh | Wkh | Wvh | Woh | Q | K | Vt | Xc  (~150 MB)
// ---------------------------------------------------------------------------
extern "C" void kernel_launch(void* const* d_in, const int* in_sizes, int n_in,
                              void* d_out, int out_size, void* d_ws, size_t ws_size,
                              hipStream_t stream) {
  const float* D     = (const float*)d_in[0];
  const float* Wq    = (const float*)d_in[1];
  const float* Wk    = (const float*)d_in[2];
  const float* Wv    = (const float*)d_in[3];
  const float* Wo    = (const float*)d_in[4];
  const float* bo    = (const float*)d_in[5];
  const float* gamma = (const float*)d_in[6];
  const float* beta  = (const float*)d_in[7];
  float* out = (float*)d_out;

  _Float16* ws = (_Float16*)d_ws;
  size_t off = 0;
  const size_t nD  = (size_t)BB * SS * DIN;        // 2,097,152
  const size_t nW  = (size_t)HH * DOUT * DIN;      // 2,097,152
  const size_t nWo = (size_t)DOUT * HD;            // 2,097,152
  const size_t nQ  = (size_t)BB * HH * SS * DOUT;  // 16,777,216

  _Float16* Dh  = ws + off; off += nD;
  _Float16* Wqh = ws + off; off += nW;
  _Float16* Wkh = ws + off; off += nW;
  _Float16* Wvh = ws + off; off += nW;
  _Float16* Woh = ws + off; off += nWo;
  _Float16* Qd  = ws + off; off += nQ;
  _Float16* Kd  = ws + off; off += nQ;
  _Float16* Vt  = ws + off; off += nQ;
  _Float16* Xc  = ws + off; off += nQ;

  cast_f32_f16<<<1024, 256, 0, stream>>>(D,  Dh,  (int)nD);
  cast_f32_f16<<<1024, 256, 0, stream>>>(Wq, Wqh, (int)nW);
  cast_f32_f16<<<1024, 256, 0, stream>>>(Wk, Wkh, (int)nW);
  cast_f32_f16<<<1024, 256, 0, stream>>>(Wv, Wvh, (int)nW);
  cast_f32_f16<<<1024, 256, 0, stream>>>(Wo, Woh, (int)nWo);

  qkv_proj<<<dim3(SS / 128, BB * HH), 256, 0, stream>>>(Dh, Wqh, Wkh, Wvh, Qd, Kd, Vt);
  attn_flash<<<dim3(SS / 16, BB * HH), 256, 0, stream>>>(Qd, Kd, Vt, Xc);
  oproj_ln<<<(BB * SS) / 16, 256, 0, stream>>>(Xc, Woh, D, bo, gamma, beta, out);
}
